// GatedSlotAttention_89584427860538
// MI455X (gfx1250) — matrix operand order repaired and checked by reading, not compiled, not run on hardware
//
#include <hip/hip_runtime.h>
#include <hip/hip_bf16.h>

// ---------------------------------------------------------------------------
// Gated Slot Attention forward for MI455X (gfx1250, wave32, WMMA).
//   1) f32->bf16 weight conversion
//   2) RMSNorm(hidden) -> bf16
//   3) 4x WMMA GEMM projections (128x128 tile, static ping-pong double buffer)
//   4) chunked (C=64) scan: recurrence as bf16 WMMA GEMMs; fp32 master state
//      in global (L2-resident); transposed bf16 LDS copies so all hot
//      fragments load as ds_load_b128
//   5) swish+RMSNorm(o) -> bf16, final WMMA GEMM -> fp32 out
// ---------------------------------------------------------------------------

typedef __attribute__((ext_vector_type(16))) __bf16 v16bf;
typedef __attribute__((ext_vector_type(8)))  __bf16 v8bf;
typedef __attribute__((ext_vector_type(8)))  float  v8f;
typedef __attribute__((ext_vector_type(4)))  float  v4f;
typedef __attribute__((ext_vector_type(4)))  unsigned int v4u;

#define DEV __device__ __forceinline__

DEV v16bf cat8(v8bf lo, v8bf hi) {
  return __builtin_shufflevector(lo, hi, 0,1,2,3,4,5,6,7,8,9,10,11,12,13,14,15);
}

DEV v8f wmma_bf16(v16bf a, v16bf b, v8f c) {
  return __builtin_amdgcn_wmma_f32_16x16x32_bf16(false, a, false, b, (short)0, c,
                                                 false, false);
}

DEV v8f vzero8() { v8f z = {0.f,0.f,0.f,0.f,0.f,0.f,0.f,0.f}; return z; }

// A-fragment (16x32 bf16), row-major source s[row*stride + col].
// Contiguous k order used here (B matches): row = lane&15; K(e) = e + 16*(lane>>4).
DEV v16bf ldsA(const __bf16* s, int stride, int r0, int kb, int lane) {
  const int half = lane >> 4;
  const __bf16* p = s + (size_t)(r0 + (lane & 15)) * stride + kb + half * 16;
  v8bf lo = *(const v8bf*)(p);
  v8bf hi = *(const v8bf*)(p + 8);
  return cat8(lo, hi);
}

// A-fragment of the transpose of a row-major source: element (r,k) = s[k][r].
DEV v16bf ldsA_trans(const __bf16* s, int stride, int r0, int kb, int lane) {
  const int half = lane >> 4, r = r0 + (lane & 15);
  v16bf a;
#pragma unroll
  for (int e = 0; e < 16; ++e) {
    int k = e + half * 16;
    a[e] = s[(size_t)(kb + k) * stride + r];
  }
  return a;
}

// B-fragment (32x16 bf16), element (k,c) = s[(kb+k)*stride + c0 + c]. Scalar.
DEV v16bf ldsB_kmajor(const __bf16* s, int stride, int kb, int c0, int lane) {
  const int half = lane >> 4, c = c0 + (lane & 15);
  v16bf b;
#pragma unroll
  for (int e = 0; e < 16; ++e) b[e] = s[(size_t)(kb + e + 16 * half) * stride + c];
  return b;
}

// B-fragment, element (k,c) = s[(c0+c)*stride + kb + k]  (K contiguous -> b128).
DEV v16bf ldsB_cmajor(const __bf16* s, int stride, int kb, int c0, int lane) {
  const int half = lane >> 4;
  const __bf16* p = s + (size_t)(c0 + (lane & 15)) * stride + kb + half * 16;
  v8bf lo = *(const v8bf*)(p);
  v8bf hi = *(const v8bf*)(p + 8);
  return cat8(lo, hi);
}

DEV float swishf(float x) { return x / (1.0f + __expf(-x)); }

// ---------------------------------------------------------------------------
__global__ void k_f32_to_bf16(const float* __restrict__ in,
                              __bf16* __restrict__ out, int n4) {
  int i = blockIdx.x * blockDim.x + threadIdx.x;
  if (i >= n4) return;
  v4f x = *(const v4f*)(in + (size_t)i * 4);
  __bf16* o = out + (size_t)i * 4;
  o[0] = (__bf16)x[0]; o[1] = (__bf16)x[1];
  o[2] = (__bf16)x[2]; o[3] = (__bf16)x[3];
}

// ---------------------------------------------------------------------------
__global__ __launch_bounds__(256)
void k_rmsnorm_bf16(const float* __restrict__ in, const float* __restrict__ wgt,
                    __bf16* __restrict__ out, int do_swish) {
  __shared__ float red[8];
  const int row = blockIdx.x, tid = threadIdx.x;
  v4f x = *(const v4f*)(in + (size_t)row * 1024 + tid * 4);
  if (do_swish) {
#pragma unroll
    for (int i = 0; i < 4; ++i) x[i] = swishf(x[i]);
  }
  float ss = x[0]*x[0] + x[1]*x[1] + x[2]*x[2] + x[3]*x[3];
#pragma unroll
  for (int off = 16; off; off >>= 1) ss += __shfl_xor(ss, off, 32);
  if ((tid & 31) == 0) red[tid >> 5] = ss;
  __syncthreads();
  if (tid < 8) {
    float v = red[tid];
#pragma unroll
    for (int off = 4; off; off >>= 1) v += __shfl_xor(v, off, 8);
    if (tid == 0) red[0] = v;
  }
  __syncthreads();
  const float scale = rsqrtf(red[0] * (1.0f / 1024.0f) + 1e-5f);
  v4f w4 = *(const v4f*)(wgt + tid * 4);
  __bf16* o = out + (size_t)row * 1024 + tid * 4;
#pragma unroll
  for (int i = 0; i < 4; ++i) o[i] = (__bf16)(x[i] * scale * w4[i]);
}

// ---------------------------------------------------------------------------
// Y = act(X[8192x1024] @ W[1024x1024]^T).  Tile 128x128, 8 waves.
// Static ping-pong double buffering: no branch in the steady-state loop so
// accumulators stay pinned (no v_dual_mov shuffling between parities).
// act: 0=none, 1=swish, 2=logsigmoid/16.
// ---------------------------------------------------------------------------
__global__ __launch_bounds__(256)
void k_gemm_bf16(const __bf16* __restrict__ X, const __bf16* __restrict__ Wt,
                 int act, int out_is_bf16,
                 __bf16* __restrict__ outb, float* __restrict__ outf) {
  __shared__ __bf16 sX[2][128 * 40];
  __shared__ __bf16 sW[2][128 * 40];
  const int tid = threadIdx.x, lane = tid & 31, w = tid >> 5;
  const int m0 = blockIdx.y * 128, n0 = blockIdx.x * 128;

  auto stage = [&](int kk, int p) {
    const int r = tid >> 1, ch = (tid & 1) * 16;
    const __bf16* g = X + (size_t)(m0 + r) * 1024 + kk + ch;
    __builtin_prefetch(g + 128, 0, 1);          // global_prefetch_b8
    *(v4u*)(&sX[p][r * 40 + ch])     = *(const v4u*)(g);
    *(v4u*)(&sX[p][r * 40 + ch + 8]) = *(const v4u*)(g + 8);
    const __bf16* g2 = Wt + (size_t)(n0 + r) * 1024 + kk + ch;
    *(v4u*)(&sW[p][r * 40 + ch])     = *(const v4u*)(g2);
    *(v4u*)(&sW[p][r * 40 + ch + 8]) = *(const v4u*)(g2 + 8);
  };

  v8f acc[8];
#pragma unroll
  for (int j = 0; j < 8; ++j) acc[j] = vzero8();

  auto compute = [&](int p) {
    v16bf a = ldsA(sX[p], 40, 16 * w, 0, lane);   // one A-frag feeds 8 WMMAs
#pragma unroll
    for (int j = 0; j < 8; ++j) {
      v16bf b = ldsB_cmajor(sW[p], 40, 0, 16 * j, lane);
      acc[j] = wmma_bf16(a, b, acc[j]);
    }
  };

  stage(0, 0);
  __syncthreads();
#pragma unroll 1
  for (int it = 0; it < 15; ++it) {
    stage(32 * (2 * it + 1), 1);  compute(0);  __syncthreads();
    stage(32 * (2 * it + 2), 0);  compute(1);  __syncthreads();
  }
  stage(31 * 32, 1);  compute(0);  __syncthreads();
  compute(1);

  const int half = lane >> 4;
#pragma unroll
  for (int j = 0; j < 8; ++j) {
#pragma unroll
    for (int e = 0; e < 8; ++e) {
      float x = acc[j][e];
      if (act == 1)      x = swishf(x);
      else if (act == 2) x = (fminf(x, 0.0f) - log1pf(__expf(-fabsf(x)))) * (1.0f / 16.0f);
      size_t row = (size_t)(m0 + 16 * w + e + 8 * half);
      size_t col = (size_t)(n0 + 16 * j + (lane & 15));
      if (out_is_bf16) outb[row * 1024 + col] = (__bf16)x;
      else             outf[row * 1024 + col] = x;
    }
  }
}

// ---------------------------------------------------------------------------
// Scan kernel: one block per (b,h); 512 threads = 16 waves; chunk C = 64.
// LDS layouts chosen so every hot B-operand is K-contiguous (ds_load_b128):
//   sSTt: Hk^T[m][dk]  then  Hv^T[dv][m]      (256 x 264)
//   sSBt: Sbar^T[m][t]                        (256 x 72)
//   sKU : K[t][dk] (pass 1)  ->  V^T[dv][t]   (union, 36864 B)
// ---------------------------------------------------------------------------
__global__ __launch_bounds__(512)
void k_gsa_scan(const __bf16* __restrict__ Qg, const __bf16* __restrict__ Kg,
                const __bf16* __restrict__ Vg, const float* __restrict__ Fg,
                float* __restrict__ Og, float* __restrict__ HkG,
                float* __restrict__ HvG) {
  constexpr int SS = 264;
  constexpr int SC = 264;
  constexpr int SB = 72;
  constexpr int SA = 72;

  __shared__ __attribute__((aligned(16))) unsigned char smem[322560];
  __bf16* sSTt = (__bf16*)smem;                  // 256 x SS
  __bf16* sQ   = sSTt + 256 * SS;                // 64 x SC
  __bf16* sKU  = sQ   + 64 * SC;                 // K: 64 x SC | V^T: 256 x SB
  __bf16* sPT  = sKU  + 256 * SB;                // 64 x SC
  __bf16* sGam = sPT  + 64 * SC;                 // 64 x SC   (exp(b))
  __bf16* sSBt = sGam + 64 * SC;                 // 256 x SB
  __bf16* sAW  = sSBt + 256 * SB;                // 64 x SA
  float*  sGC  = (float*)(sAW + 64 * SA);        // 256  (exp(b_63[m]))
  float*  sRA  = sGC + 256;                      // 64 x 4 (max partials)
  float*  sRB  = sRA + 256;                      // 64 x 4 (sum partials)

  const int tid = threadIdx.x, lane = tid & 31, w = tid >> 5;
  const int bh = blockIdx.x, b = bh >> 2, h = bh & 3;
  const size_t seqBase = ((size_t)b * 2048) * 1024 + (size_t)h * 256;
  float* hk = HkG + (size_t)bh * 65536;
  float* hv = HvG + (size_t)bh * 65536;

  {  // zero fp32 master states (deterministic per launch)
    v4f z = {0.f, 0.f, 0.f, 0.f};
    for (int i = tid * 4; i < 65536; i += 512 * 4) {
      *(v4f*)(hk + i) = z;
      *(v4f*)(hv + i) = z;
    }
  }
  __threadfence_block();
  __syncthreads();

  for (int t0 = 0; t0 < 2048; t0 += 64) {
    // ---- Phase 1: stage Q,K; materialize Hk^T; gate cumsum ----
    {
      int r = tid >> 3, c = (tid & 7) * 32;
      const __bf16* gq = Qg + seqBase + (size_t)(t0 + r) * 1024 + c;
      const __bf16* gk = Kg + seqBase + (size_t)(t0 + r) * 1024 + c;
      __builtin_prefetch(gq, 0, 1);
#pragma unroll
      for (int u = 0; u < 4; ++u) {
        *(v4u*)(&sQ[r * SC + c + u * 8])  = *(const v4u*)(gq + u * 8);
        *(v4u*)(&sKU[r * SC + c + u * 8]) = *(const v4u*)(gk + u * 8);
      }
    }
    for (int i = tid; i < 16384; i += 512) {   // Hk^T[m][dk] <- hk[dk][m]
      int m = i & 255, c0 = (i >> 8) * 4;
      __bf16* d = &sSTt[(size_t)m * SS + c0];
#pragma unroll
      for (int u = 0; u < 4; ++u) d[u] = (__bf16)hk[(size_t)(c0 + u) * 256 + m];
    }
    if (tid < 256) {  // per-slot cumulative log-gate
      const int m = tid;
      float bacc = 0.f;
      for (int t = 0; t < 64; ++t) {
        float g = Fg[seqBase + (size_t)(t0 + t) * 1024 + m];
        bacc += g;
        float eb = __expf(bacc);
        sGam[t * SC + m] = (__bf16)eb;                              // Gamma
        sSBt[m * SB + t] = (__bf16)((1.0f - __expf(g)) / eb);       // Sbar^T
        if (t == 63) sGC[m] = eb;
      }
    }
    __syncthreads();

    // ---- 2a: A = tril(Q K^T) ----
    {
      const int ti = w >> 2, tj = w & 3;
      v8f acc = vzero8();
      for (int kk = 0; kk < 256; kk += 32) {
        v16bf a = ldsA(sQ, SC, 16 * ti, kk, lane);
        v16bf bb = ldsB_cmajor(sKU, SC, kk, 16 * tj, lane);
        acc = wmma_bf16(a, bb, acc);
      }
      const int cl = 16 * tj + (lane & 15);
#pragma unroll
      for (int e = 0; e < 8; ++e) {
        int rg = 16 * ti + e + 8 * (lane >> 4);
        sAW[rg * SA + cl] = (__bf16)((cl <= rg) ? acc[e] : 0.0f);
      }
    }
    __syncthreads();

    // ---- 2b: z = Gamma * (Q @ Hk0 + A @ Sbar)   (kept in registers) ----
    const int r = w & 3, jb = w >> 2;
    v8f zreg[4];
#pragma unroll
    for (int jj = 0; jj < 4; ++jj) zreg[jj] = vzero8();
    for (int kk = 0; kk < 256; kk += 32) {
      v16bf a = ldsA(sQ, SC, 16 * r, kk, lane);
#pragma unroll
      for (int jj = 0; jj < 4; ++jj) {
        v16bf bb = ldsB_cmajor(sSTt, SS, kk, 16 * (jb + 4 * jj), lane);
        zreg[jj] = wmma_bf16(a, bb, zreg[jj]);
      }
    }
#pragma unroll
    for (int tt = 0; tt < 64; tt += 32) {
      v16bf a = ldsA(sAW, SA, 16 * r, tt, lane);
#pragma unroll
      for (int jj = 0; jj < 4; ++jj) {
        v16bf bb = ldsB_cmajor(sSBt, SB, tt, 16 * (jb + 4 * jj), lane);
        zreg[jj] = wmma_bf16(a, bb, zreg[jj]);
      }
    }
#pragma unroll
    for (int jj = 0; jj < 4; ++jj) {
      const int cl = 16 * (jb + 4 * jj) + (lane & 15);
#pragma unroll
      for (int e = 0; e < 8; ++e) {
        int tg = 16 * r + e + 8 * (lane >> 4);
        zreg[jj][e] *= (float)sGam[tg * SC + cl];
      }
    }

    // ---- 2d: Hk <- Gamma_C * (Hk + K^T @ Sbar)  (fp32 RMW in global) ----
    {
      v16bf at0 = ldsA_trans(sKU, SC, 16 * w, 0, lane);
      v16bf at1 = ldsA_trans(sKU, SC, 16 * w, 32, lane);
      for (int j = 0; j < 16; ++j) {
        v8f acc = vzero8();
        acc = wmma_bf16(at0, ldsB_cmajor(sSBt, SB, 0, 16 * j, lane), acc);
        acc = wmma_bf16(at1, ldsB_cmajor(sSBt, SB, 32, 16 * j, lane), acc);
        const int m = 16 * j + (lane & 15);
        const float gc = sGC[m];
#pragma unroll
        for (int e = 0; e < 8; ++e) {
          int dk = 16 * w + e + 8 * (lane >> 4);
          size_t idx = (size_t)dk * 256 + m;
          hk[idx] = (hk[idx] + acc[e]) * gc;
        }
      }
    }
    __syncthreads();

    // ---- softmax over slots (from registers; cross-wave 64x4 partials) ----
    {
      const int q = w >> 2;
#pragma unroll
      for (int e = 0; e < 8; ++e) {
        float m0 = fmaxf(fmaxf(zreg[0][e], zreg[1][e]),
                         fmaxf(zreg[2][e], zreg[3][e]));
#pragma unroll
        for (int off = 8; off; off >>= 1) m0 = fmaxf(m0, __shfl_xor(m0, off, 16));
        if ((lane & 15) == 0) {
          int tg = 16 * r + e + 8 * (lane >> 4);
          sRA[tg * 4 + q] = m0;
        }
      }
      __syncthreads();
#pragma unroll
      for (int e = 0; e < 8; ++e) {
        int tg = 16 * r + e + 8 * (lane >> 4);
        float gm = fmaxf(fmaxf(sRA[tg * 4 + 0], sRA[tg * 4 + 1]),
                         fmaxf(sRA[tg * 4 + 2], sRA[tg * 4 + 3]));
        float s = 0.f;
#pragma unroll
        for (int jj = 0; jj < 4; ++jj) {
          zreg[jj][e] = __expf(zreg[jj][e] - gm);
          s += zreg[jj][e];
        }
#pragma unroll
        for (int off = 8; off; off >>= 1) s += __shfl_xor(s, off, 16);
        if ((lane & 15) == 0) sRB[tg * 4 + q] = s;   // separate buffer: no race
      }
      __syncthreads();
    }
    {  // combine sums; write P~ = softmax(z) * Gamma
#pragma unroll
      for (int jj = 0; jj < 4; ++jj) {
        const int cl = 16 * (jb + 4 * jj) + (lane & 15);
#pragma unroll
        for (int e = 0; e < 8; ++e) {
          int tg = 16 * r + e + 8 * (lane >> 4);
          float gs = (sRB[tg * 4 + 0] + sRB[tg * 4 + 1]) +
                     (sRB[tg * 4 + 2] + sRB[tg * 4 + 3]);
          float p = zreg[jj][e] / gs;
          sPT[tg * SC + cl] = (__bf16)(p * (float)sGam[tg * SC + cl]);
        }
      }
    }
    {  // stage V^T[dv][t]  (K fully consumed in 2d; barrier passed)
      int t = tid >> 3, dvb = (tid & 7) * 32;
      const __bf16* gv = Vg + seqBase + (size_t)(t0 + t) * 1024 + dvb;
#pragma unroll
      for (int c = 0; c < 4; ++c) {
        v8bf d = *(const v8bf*)(gv + c * 8);
#pragma unroll
        for (int u = 0; u < 8; ++u) sKU[(dvb + c * 8 + u) * SB + t] = d[u];
      }
    }
    for (int i = tid; i < 16384; i += 512) {   // Hv^T[dv][m] <- hv[m][dv]
      int dv = i & 255, c0 = (i >> 8) * 4;
      __bf16* d = &sSTt[(size_t)dv * SS + c0];
#pragma unroll
      for (int u = 0; u < 4; ++u) d[u] = (__bf16)hv[(size_t)(c0 + u) * 256 + dv];
    }
    __syncthreads();

    // ---- W = tril(P~ @ Sbar^T) ----
    {
      const int ti = w >> 2, tj = w & 3;
      v8f acc = vzero8();
      for (int mm = 0; mm < 256; mm += 32) {
        v16bf a = ldsA(sPT, SC, 16 * ti, mm, lane);
        v16bf bb = ldsB_kmajor(sSBt, SB, mm, 16 * tj, lane);
        acc = wmma_bf16(a, bb, acc);
      }
      const int cl = 16 * tj + (lane & 15);
#pragma unroll
      for (int e = 0; e < 8; ++e) {
        int rg = 16 * ti + e + 8 * (lane >> 4);
        sAW[rg * SA + cl] = (__bf16)((cl <= rg) ? acc[e] : 0.0f);
      }
    }
    __syncthreads();

    // ---- O = P~ @ Hv0 + W @ V  -> global fp32 ----
    {
      v8f oacc[4];
#pragma unroll
      for (int jj = 0; jj < 4; ++jj) oacc[jj] = vzero8();
      for (int kk = 0; kk < 256; kk += 32) {
        v16bf a = ldsA(sPT, SC, 16 * r, kk, lane);
#pragma unroll
        for (int jj = 0; jj < 4; ++jj) {
          v16bf bb = ldsB_cmajor(sSTt, SS, kk, 16 * (jb + 4 * jj), lane);
          oacc[jj] = wmma_bf16(a, bb, oacc[jj]);
        }
      }
#pragma unroll
      for (int tt = 0; tt < 64; tt += 32) {
        v16bf a = ldsA(sAW, SA, 16 * r, tt, lane);
#pragma unroll
        for (int jj = 0; jj < 4; ++jj) {
          v16bf bb = ldsB_cmajor(sKU, SB, tt, 16 * (jb + 4 * jj), lane);
          oacc[jj] = wmma_bf16(a, bb, oacc[jj]);
        }
      }
#pragma unroll
      for (int jj = 0; jj < 4; ++jj) {
        const int dv = 16 * (jb + 4 * jj) + (lane & 15);
#pragma unroll
        for (int e = 0; e < 8; ++e) {
          int tg = 16 * r + e + 8 * (lane >> 4);
          Og[seqBase + (size_t)(t0 + tg) * 1024 + dv] = oacc[jj][e];
        }
      }
    }

    // ---- Hv <- Gamma_C * (Hv + Sbar^T @ V)  (fp32 RMW in global) ----
    {
      v16bf at0 = ldsA(sSBt, SB, 16 * w, 0, lane);
      v16bf at1 = ldsA(sSBt, SB, 16 * w, 32, lane);
      for (int j = 0; j < 16; ++j) {
        v8f acc = vzero8();
        acc = wmma_bf16(at0, ldsB_cmajor(sKU, SB, 0, 16 * j, lane), acc);
        acc = wmma_bf16(at1, ldsB_cmajor(sKU, SB, 32, 16 * j, lane), acc);
        const int dv = 16 * j + (lane & 15);
#pragma unroll
        for (int e = 0; e < 8; ++e) {
          int m = 16 * w + e + 8 * (lane >> 4);
          size_t idx = (size_t)m * 256 + dv;
          hv[idx] = (hv[idx] + acc[e]) * sGC[m];
        }
      }
    }
    __threadfence_block();
    __syncthreads();
  }
}

// ---------------------------------------------------------------------------
extern "C" void kernel_launch(void* const* d_in, const int* in_sizes, int n_in,
                              void* d_out, int out_size, void* d_ws, size_t ws_size,
                              hipStream_t stream) {
  (void)in_sizes; (void)n_in; (void)out_size; (void)ws_size;
  const float* hs = (const float*)d_in[0];
  const float* nw = (const float*)d_in[1];
  const float* wq = (const float*)d_in[2];
  const float* wk = (const float*)d_in[3];
  const float* wv = (const float*)d_in[4];
  const float* wf = (const float*)d_in[5];
  const float* gw = (const float*)d_in[6];
  const float* wo = (const float*)d_in[7];
  float* out = (float*)d_out;

  char* ws = (char*)d_ws;
  size_t off = 0;
  auto carve = [&](size_t bytes) -> void* {
    void* p = ws + off;
    off += (bytes + 255) & ~(size_t)255;
    return p;
  };
  const size_t WB  = (size_t)1024 * 1024 * 2;
  const size_t AB  = (size_t)8192 * 1024 * 2;
  const size_t AF  = (size_t)8192 * 1024 * 4;
  const size_t STB = (size_t)16 * 65536 * 4;

  __bf16* wqb = (__bf16*)carve(WB);
  __bf16* wkb = (__bf16*)carve(WB);
  __bf16* wvb = (__bf16*)carve(WB);
  __bf16* wfb = (__bf16*)carve(WB);
  __bf16* wob = (__bf16*)carve(WB);
  __bf16* hnb = (__bf16*)carve(AB);
  __bf16* qb  = (__bf16*)carve(AB);
  __bf16* kb  = (__bf16*)carve(AB);
  __bf16* vb  = (__bf16*)carve(AB);
  float*  ff  = (float*)carve(AF);
  float*  o32 = (float*)carve(AF);
  float*  gHk = (float*)carve(STB);
  float*  gHv = (float*)carve(STB);

  const int n4 = (1024 * 1024) / 4;
  k_f32_to_bf16<<<n4 / 256, 256, 0, stream>>>(wq, wqb, n4);
  k_f32_to_bf16<<<n4 / 256, 256, 0, stream>>>(wk, wkb, n4);
  k_f32_to_bf16<<<n4 / 256, 256, 0, stream>>>(wv, wvb, n4);
  k_f32_to_bf16<<<n4 / 256, 256, 0, stream>>>(wf, wfb, n4);
  k_f32_to_bf16<<<n4 / 256, 256, 0, stream>>>(wo, wob, n4);

  k_rmsnorm_bf16<<<8192, 256, 0, stream>>>(hs, nw, hnb, 0);

  dim3 gg(8, 64), gb(256);
  k_gemm_bf16<<<gg, gb, 0, stream>>>(hnb, wqb, 1, 1, qb, nullptr);
  k_gemm_bf16<<<gg, gb, 0, stream>>>(hnb, wkb, 1, 1, kb, nullptr);
  k_gemm_bf16<<<gg, gb, 0, stream>>>(hnb, wvb, 1, 1, vb, nullptr);
  k_gemm_bf16<<<gg, gb, 0, stream>>>(hnb, wfb, 2, 0, nullptr, ff);

  k_gsa_scan<<<16, 512, 0, stream>>>(qb, kb, vb, ff, o32, gHk, gHv);

  k_rmsnorm_bf16<<<8192, 256, 0, stream>>>(o32, gw, hnb, 1);
  k_gemm_bf16<<<gg, gb, 0, stream>>>(hnb, wob, 0, 0, nullptr, out);
}
